// GELU121_39857296507267
// MI455X (gfx1250) — matrix operand order repaired and checked by reading, not compiled, not run on hardware
//
#include <hip/hip_runtime.h>
#include <hip/hip_bf16.h>
#include <math.h>

// Problem constants (from reference): B=4, T=4096, D=512, fp32 in/out.
#define BB 4
#define TT 4096
#define DD 512
#define RT  (TT / 16)              // 256 16-row tiles per batch
#define RB  (TT / 32)              // 128 32-row blocks per batch (2x2 register blocking)
#define COL_CHUNKS 4
#define TILES_PER_CHUNK (RT / COL_CHUNKS)   // 64 column tiles per chunk (even)

typedef __attribute__((ext_vector_type(16))) _Float16 v16h;
typedef __attribute__((ext_vector_type(8)))  _Float16 v8h;
typedef __attribute__((ext_vector_type(8)))  float    v8f;

// Order-preserving float <-> int key (involution); valid for non-NaN floats.
__device__ __forceinline__ int f32key(float f) {
    int i = __float_as_int(f);
    return i ^ ((i >> 31) & 0x7fffffff);
}
__device__ __forceinline__ float keyf32(int k) {
    return __int_as_float(k ^ ((k >> 31) & 0x7fffffff));
}

// WMMA fragment loaders (16-bit layouts per CDNA5 ISA 7.12.2).
// A 16x32: lane l holds row (l&15), K = [hi*8 .. +7] and [16+hi*8 .. +7].
__device__ __forceinline__ v16h load_afrag(const _Float16* __restrict__ arow, int ks, int hi) {
    union { v16h v; v8h h[2]; } a;
    a.h[0] = *(const v8h*)(arow + ks + hi * 8);
    a.h[1] = *(const v8h*)(arow + ks + 16 + hi * 8);
    return a.v;
}
// B 32x16: lane l holds col (l&15) == row of X̂, K = [hi*8 .. +7] and [16+hi*8 .. +7].
__device__ __forceinline__ v16h load_bfrag(const _Float16* __restrict__ brow, int ks, int hi) {
    union { v16h v; v8h h[2]; } b;
    b.h[0] = *(const v8h*)(brow + ks + hi * 8);
    b.h[1] = *(const v8h*)(brow + ks + 16 + hi * 8);
    return b.v;
}
__device__ __forceinline__ v8f wmma16(v16h a, v16h b, v8f c) {
    return __builtin_amdgcn_wmma_f32_16x16x32_f16(false, a, false, b, (short)0, c,
                                                  false, false);
}

// ---------------------------------------------------------------------------
// Kernel 1: per-row L2 norm, pack normalized rows to f16, init rowmax keys.
// One wave per row (D=512 -> 16 elements/lane).
// ---------------------------------------------------------------------------
__global__ __launch_bounds__(256)
void norm_pack_kernel(const float* __restrict__ x,
                      _Float16* __restrict__ xh,
                      int* __restrict__ rowkey) {
    const int wave = threadIdx.x >> 5;
    const int lane = threadIdx.x & 31;
    const int row  = blockIdx.x * 8 + wave;           // 0 .. B*T-1

    const float* xr = x + (size_t)row * DD + lane * 16;
    float4 q[4];
#pragma unroll
    for (int i = 0; i < 4; ++i) q[i] = ((const float4*)xr)[i];

    float ss = 0.0f;
#pragma unroll
    for (int i = 0; i < 4; ++i)
        ss += q[i].x * q[i].x + q[i].y * q[i].y + q[i].z * q[i].z + q[i].w * q[i].w;

#pragma unroll
    for (int m = 16; m >= 1; m >>= 1) ss += __shfl_xor(ss, m, 32);

    const float inv = 1.0f / fmaxf(sqrtf(ss), 1e-12f);

    _Float16 hv[16];
#pragma unroll
    for (int i = 0; i < 4; ++i) {
        hv[4 * i + 0] = (_Float16)(q[i].x * inv);
        hv[4 * i + 1] = (_Float16)(q[i].y * inv);
        hv[4 * i + 2] = (_Float16)(q[i].z * inv);
        hv[4 * i + 3] = (_Float16)(q[i].w * inv);
    }
    v8h* dst = (v8h*)(xh + (size_t)row * DD + lane * 16);
    dst[0] = *(const v8h*)(hv);
    dst[1] = *(const v8h*)(hv + 8);

    if (lane == 0) rowkey[row] = f32key(-2.0f);       // masked-everything value
}

// ---------------------------------------------------------------------------
// Kernel 2: causal row-max of X̂·X̂ᵀ via V_WMMA_F32_16X16X32_F16.
// 2x2 register blocking: one wave per (batch, 32-row block, column chunk).
// 4 fragment loads -> 4 WMMAs per K-step (1 VMEM per WMMA).
// ---------------------------------------------------------------------------
__global__ __launch_bounds__(32)
void sim_rowmax_kernel(const _Float16* __restrict__ xh,
                       int* __restrict__ rowkey) {
    const int lane = threadIdx.x;          // 0..31, all active
    const int m16  = lane & 15;
    const int hi   = lane >> 4;

    const int b   = blockIdx.x / RB;
    const int jb  = blockIdx.x % RB;       // 32-row block index
    const int jt0 = 2 * jb;                // first 16-row tile
    const int jt1 = 2 * jb + 1;            // second 16-row tile
    const int R0  = jb * 32;

    const int ctLo = blockIdx.y * TILES_PER_CHUNK;
    const int ctHi = min(ctLo + TILES_PER_CHUNK, jt1 + 1);
    if (ctLo >= ctHi) return;              // wave-uniform -> EXEC stays all-ones

    const _Float16* xb = xh + (size_t)b * TT * DD;
    const _Float16* arow0 = xb + (size_t)(R0 + m16) * DD;        // rows R0..R0+15
    const _Float16* arow1 = arow0 + (size_t)16 * DD;             // rows R0+16..R0+31

    v8f mx0 = {-2.0f, -2.0f, -2.0f, -2.0f, -2.0f, -2.0f, -2.0f, -2.0f};
    v8f mx1 = mx0;
    const v8f zero = {0.0f, 0.0f, 0.0f, 0.0f, 0.0f, 0.0f, 0.0f, 0.0f};

    // ---- Paired full tiles: ct, ct+1 both strictly below the diagonal block.
    // ctLo and jt0 are both even -> count of full tiles in chunk is even.
    const int ctFullEnd = min(ctHi, jt0);
    for (int ct = ctLo; ct + 2 <= ctFullEnd; ct += 2) {
        const _Float16* b0row = xb + (size_t)(ct * 16 + m16) * DD;
        const _Float16* b1row = b0row + (size_t)16 * DD;
        v8f a00 = zero, a01 = zero, a10 = zero, a11 = zero;
#pragma unroll 2
        for (int ks = 0; ks < DD; ks += 32) {
            const v16h A0 = load_afrag(arow0, ks, hi);
            const v16h A1 = load_afrag(arow1, ks, hi);
            const v16h B0 = load_bfrag(b0row, ks, hi);
            const v16h B1 = load_bfrag(b1row, ks, hi);
            a00 = wmma16(A0, B0, a00);
            a01 = wmma16(A0, B1, a01);
            a10 = wmma16(A1, B0, a10);
            a11 = wmma16(A1, B1, a11);
        }
#pragma unroll
        for (int r = 0; r < 8; ++r) {
            mx0[r] = fmaxf(mx0[r], fmaxf(a00[r], a01[r]));
            mx1[r] = fmaxf(mx1[r], fmaxf(a10[r], a11[r]));
        }
    }

    // ---- Boundary tile ct == jt0: diagonal for row tile 0, fully valid for tile 1.
    if (jt0 >= ctLo && jt0 < ctHi) {
        const _Float16* brow = xb + (size_t)(jt0 * 16 + m16) * DD;  // cols R0..R0+15
        v8f a0 = zero, a1 = zero;
#pragma unroll 4
        for (int ks = 0; ks < DD; ks += 32) {
            const v16h A0 = load_afrag(arow0, ks, hi);
            const v16h A1 = load_afrag(arow1, ks, hi);
            const v16h B  = load_bfrag(brow,  ks, hi);
            a0 = wmma16(A0, B, a0);
            a1 = wmma16(A1, B, a1);
        }
#pragma unroll
        for (int r = 0; r < 8; ++r) {
            const bool keep = m16 < (hi * 8 + r);   // col R0+m16 < row R0+hi*8+r
            mx0[r] = fmaxf(mx0[r], keep ? a0[r] : -2.0f);
            mx1[r] = fmaxf(mx1[r], a1[r]);          // rows >= R0+16 > all cols here
        }
    }

    // ---- Boundary tile ct == jt1: diagonal for row tile 1, fully masked for tile 0.
    if (jt1 >= ctLo && jt1 < ctHi) {
        const _Float16* brow = xb + (size_t)(jt1 * 16 + m16) * DD;  // cols R0+16..R0+31
        v8f a1 = zero;
#pragma unroll 4
        for (int ks = 0; ks < DD; ks += 32) {
            const v16h A1 = load_afrag(arow1, ks, hi);
            const v16h B  = load_bfrag(brow,  ks, hi);
            a1 = wmma16(A1, B, a1);
        }
#pragma unroll
        for (int r = 0; r < 8; ++r) {
            const bool keep = m16 < (hi * 8 + r);
            mx1[r] = fmaxf(mx1[r], keep ? a1[r] : -2.0f);
        }
    }

    // ---- Max across the 16 lanes of each half (columns); rows live in VGPR index.
#pragma unroll
    for (int r = 0; r < 8; ++r) {
        float v0 = mx0[r], v1 = mx1[r];
        v0 = fmaxf(v0, __shfl_xor(v0, 1, 32));
        v1 = fmaxf(v1, __shfl_xor(v1, 1, 32));
        v0 = fmaxf(v0, __shfl_xor(v0, 2, 32));
        v1 = fmaxf(v1, __shfl_xor(v1, 2, 32));
        v0 = fmaxf(v0, __shfl_xor(v0, 4, 32));
        v1 = fmaxf(v1, __shfl_xor(v1, 4, 32));
        v0 = fmaxf(v0, __shfl_xor(v0, 8, 32));
        v1 = fmaxf(v1, __shfl_xor(v1, 8, 32));
        mx0[r] = v0;
        mx1[r] = v1;
    }

    if (m16 == 0) {                        // lanes 0 (M=0-7) and 16 (M=8-15)
        int* rk = rowkey + b * TT;
#pragma unroll
        for (int r = 0; r < 8; ++r) {
            atomicMax(rk + R0 + hi * 8 + r,      f32key(mx0[r]));
            atomicMax(rk + R0 + 16 + hi * 8 + r, f32key(mx1[r]));
        }
    }
}

// ---------------------------------------------------------------------------
// Kernel 3: gate = 1 + softplus(log_alpha) * novelty(rowmax); out = gelu(x*gate)
// ---------------------------------------------------------------------------
__device__ __forceinline__ float gelu_tanh(float y) {
    const float c = 0.7978845608028654f;   // sqrt(2/pi)
    return 0.5f * y * (1.0f + tanhf(c * (y + 0.044715f * y * y * y)));
}

__global__ __launch_bounds__(256)
void gate_gelu_kernel(const float* __restrict__ x,
                      const float* __restrict__ log_alpha,
                      const int* __restrict__ rowkey,
                      float* __restrict__ out) {
    const size_t i4 = (size_t)blockIdx.x * blockDim.x + threadIdx.x;
    const size_t n4 = (size_t)BB * TT * DD / 4;
    if (i4 >= n4) return;

    const float la    = log_alpha[0];
    const float alpha = (la > 20.0f) ? la : log1pf(expf(la));   // softplus

    const size_t row = (i4 * 4) / DD;      // D divisible by 4 -> whole float4 same row
    float ms = keyf32(rowkey[row]);
    ms = fmaxf(ms, -1.0f);
    const float familiarity = (ms + 1.0f) * 0.5f;
    const float gate = 1.0f + alpha * (1.0f - familiarity);

    float4 v = ((const float4*)x)[i4];
    float4 g;
    g.x = gelu_tanh(v.x * gate);
    g.y = gelu_tanh(v.y * gate);
    g.z = gelu_tanh(v.z * gate);
    g.w = gelu_tanh(v.w * gate);
    ((float4*)out)[i4] = g;
}

// ---------------------------------------------------------------------------
extern "C" void kernel_launch(void* const* d_in, const int* in_sizes, int n_in,
                              void* d_out, int out_size, void* d_ws, size_t ws_size,
                              hipStream_t stream) {
    const float* x         = (const float*)d_in[0];
    const float* log_alpha = (const float*)d_in[1];
    float* out             = (float*)d_out;

    // Workspace layout: [0, 16MB) packed f16 normalized X; then B*T int rowmax keys.
    _Float16* xh = (_Float16*)d_ws;
    int* rowkey  = (int*)((char*)d_ws + (size_t)BB * TT * DD * sizeof(_Float16));

    // K1: norms + f16 pack + rowmax init. One wave per row, 8 waves/block.
    norm_pack_kernel<<<(BB * TT) / 8, 256, 0, stream>>>(x, xh, rowkey);

    // K2: WMMA similarity row-max, 2x2 register blocked.
    dim3 g2(BB * RB, COL_CHUNKS, 1);
    sim_rowmax_kernel<<<g2, 32, 0, stream>>>(xh, rowkey);

    // K3: elementwise gate + GELU, float4 per thread.
    const int n4 = BB * TT * DD / 4;
    gate_gelu_kernel<<<(n4 + 255) / 256, 256, 0, stream>>>(x, log_alpha, rowkey, out);
}
